// CRF_61753039782392
// MI455X (gfx1250) — compile-verified, matrix operand order checked by reading, not run on hardware
//
#include <hip/hip_runtime.h>

typedef __attribute__((ext_vector_type(16))) _Float16 v16h;
typedef __attribute__((ext_vector_type(8)))  float    v8f;

#define Bc 512
#define Sc 512
#define Lc 48
#define STARTc 46
#define PADc 47
#define STR 72   /* Ph row stride in halves: 144 B, 16B-aligned, conflict-free b128 */

// ---------------------------------------------------------------------------
// Prep: expTh[64][48] = f16(exp(T)) with K padded to 64 (rows 48..63 = 0);
//       expTpad[48]   = exp(T[k][PAD]) in f32 for the final logZ dot.
// ---------------------------------------------------------------------------
__global__ void crf_prep(const float* __restrict__ T,
                         _Float16* __restrict__ expTh,
                         float* __restrict__ expTpad) {
  for (int i = threadIdx.x; i < 64 * Lc; i += blockDim.x) {
    int k = i / Lc, n = i % Lc;
    float v = (k < Lc) ? __expf(T[k * Lc + n]) : 0.0f;
    expTh[i] = (_Float16)v;
  }
  for (int i = threadIdx.x; i < Lc; i += blockDim.x)
    expTpad[i] = __expf(T[i * Lc + PADc]);
}

// ---------------------------------------------------------------------------
// Scaled forward algorithm, one wave per 16-batch tile.
// P (= exp(alpha - logC)) lives in LDS as f16 in the WMMA A-matrix row-major
// layout (16 rows x 64 K, zero-padded) -> A fragments are raw ds_load_b128.
// Per step: S = (P @ expT) * exp(emit_t)   [6x v_wmma_f32_16x16x32_f16]
//           rowmax via shfl_xor butterfly; logC += log(mx); P = S * rcp(mx)
//           rows with t >= length stay frozen (prefix masks -> len compare).
// ---------------------------------------------------------------------------
__global__ __launch_bounds__(32) void crf_forward(
    const float* __restrict__ emit, const unsigned char* __restrict__ masks,
    const float* __restrict__ T, const _Float16* __restrict__ expTh,
    const float* __restrict__ expTpad, float* __restrict__ logZpart) {
  __shared__ __align__(16) _Float16 Ph[16 * STR];
  __shared__ float logCrow[16];
  __shared__ float red[16];

  const int lane = threadIdx.x;
  const int row  = lane & 15;
  const int hi   = lane >> 4;
  const int base = blockIdx.x * 16;

  // Constant B fragments: expT, K padded to 64, three 16-wide N tiles.
  v16h bfrag[3][2];
#pragma unroll
  for (int t3 = 0; t3 < 3; ++t3)
#pragma unroll
    for (int kh = 0; kh < 2; ++kh)
#pragma unroll
      for (int i = 0; i < 16; ++i) {
        int koff = (i < 8 ? i : i + 8) + (hi ? 8 : 0);
        bfrag[t3][kh][i] = expTh[(32 * kh + koff) * Lc + 16 * t3 + row];
      }

  // Sequence lengths from prefix masks (one byte-sum per row, once).
  int lenR = 0;
  if (hi == 0) {
    const uint4* mp = (const uint4*)(masks + (size_t)(base + row) * Sc);
    for (int i = 0; i < Sc / 16; ++i) {
      uint4 q = mp[i];
      lenR += (int)((q.x * 0x01010101u) >> 24) + (int)((q.y * 0x01010101u) >> 24)
            + (int)((q.z * 0x01010101u) >> 24) + (int)((q.w * 0x01010101u) >> 24);
    }
  }
  int len8[8];
#pragma unroll
  for (int r = 0; r < 8; ++r) len8[r] = __shfl(lenR, r + 8 * hi, 32);

  // Init: alpha0 = emit[b,0,:] + T[START,:]; P = exp(alpha0 - max); logC = max
  float mx0 = 0.0f;
  if (hi == 0) {
    const float* e0 = emit + (size_t)(base + row) * Sc * Lc;
    float m = -3.4e38f;
    for (int k = 0; k < Lc; ++k) m = fmaxf(m, e0[k] + T[STARTc * Lc + k]);
    mx0 = m;
    for (int k = 0; k < Lc; ++k)
      Ph[row * STR + k] = (_Float16)__expf(e0[k] + T[STARTc * Lc + k] - m);
    for (int k = Lc; k < 64; ++k) Ph[row * STR + k] = (_Float16)0.0f;  // K pad
  }
  float logC8[8];
#pragma unroll
  for (int r = 0; r < 8; ++r) logC8[r] = __shfl(mx0, r + 8 * hi, 32);
  __syncthreads();

  for (int t = 1; t < Sc; ++t) {
    if (t + 1 < Sc)  // warm next step's emit rows (global_prefetch_b8)
      __builtin_prefetch(&emit[((size_t)(base + row) * Sc + (t + 1)) * Lc + hi * 24], 0, 1);

    // A fragments: raw 128-bit LDS loads, already in WMMA A layout.
    const _Float16* pr = &Ph[row * STR];
    union { v16h v; uint4 q[2]; } A0, A1;
    A0.q[0] = *(const uint4*)(pr + hi * 8);        // K =  0..7  (+8 hi)
    A0.q[1] = *(const uint4*)(pr + 16 + hi * 8);   // K = 16..23 (+8 hi)
    A1.q[0] = *(const uint4*)(pr + 32 + hi * 8);   // K = 32..39 (+8 hi)
    A1.q[1] = *(const uint4*)(pr + 48 + hi * 8);   // K = 48..55 (+8 hi) -> zeros

    v8f acc[3];
#pragma unroll
    for (int t3 = 0; t3 < 3; ++t3) {
      v8f c = {};
      c = __builtin_amdgcn_wmma_f32_16x16x32_f16(false, A0.v, false, bfrag[t3][0],
                                                 (short)0, c, false, false);
      c = __builtin_amdgcn_wmma_f32_16x16x32_f16(false, A1.v, false, bfrag[t3][1],
                                                 (short)0, c, false, false);
      acc[t3] = c;
    }

    // S = acc * exp(emit_t); C layout: M = r + 8*hi, N = 16*t3 + row
    float val[3][8];
#pragma unroll
    for (int t3 = 0; t3 < 3; ++t3) {
      int n = 16 * t3 + row;
#pragma unroll
      for (int r = 0; r < 8; ++r) {
        float e = emit[((size_t)(base + r + 8 * hi) * Sc + t) * Lc + n];
        val[t3][r] = acc[t3][r] * __expf(e);
      }
    }

    // Row maxima: local reduce over tiles + shfl_xor butterfly over 16 lanes.
    float mx[8];
#pragma unroll
    for (int r = 0; r < 8; ++r)
      mx[r] = fmaxf(fmaxf(val[0][r], val[1][r]), val[2][r]);
#pragma unroll
    for (int d = 1; d < 16; d <<= 1)
#pragma unroll
      for (int r = 0; r < 8; ++r)
        mx[r] = fmaxf(mx[r], __shfl_xor(mx[r], d, 32));

    // Renormalize + accumulate logC (rows past their length stay frozen).
    float rcpv[8];
#pragma unroll
    for (int r = 0; r < 8; ++r) {
      float lg = __logf(mx[r]);
      if (t < len8[r]) logC8[r] += lg;
      rcpv[r] = __builtin_amdgcn_rcpf(mx[r]);
    }
#pragma unroll
    for (int t3 = 0; t3 < 3; ++t3) {
      int n = 16 * t3 + row;
#pragma unroll
      for (int r = 0; r < 8; ++r)
        if (t < len8[r])
          Ph[(r + 8 * hi) * STR + n] = (_Float16)(val[t3][r] * rcpv[r]);
    }
    __syncthreads();
  }

  // Merge replicated logC to per-row, then logZ_b = logC + log(P . expTpad).
  if (lane == 0)
#pragma unroll
    for (int r = 0; r < 8; ++r) logCrow[r] = logC8[r];
  if (lane == 16)
#pragma unroll
    for (int r = 0; r < 8; ++r) logCrow[8 + r] = logC8[r];
  __syncthreads();
  if (hi == 0) {
    float s = 0.0f;
    for (int k = 0; k < Lc; ++k) s += (float)Ph[row * STR + k] * expTpad[k];
    red[row] = logCrow[row] + __logf(s);
  }
  __syncthreads();
  if (lane == 0) {
    float tot = 0.0f;
    for (int r = 0; r < 16; ++r) tot += red[r];
    logZpart[blockIdx.x] = tot;
  }
}

// ---------------------------------------------------------------------------
// Gold path score: one thread per batch (prefix masks -> early break).
// ---------------------------------------------------------------------------
__global__ void crf_gold(const float* __restrict__ emit,
                         const int* __restrict__ labels,
                         const unsigned char* __restrict__ masks,
                         const float* __restrict__ T, float* __restrict__ gold) {
  int b = blockIdx.x * blockDim.x + threadIdx.x;
  if (b >= Bc) return;
  const int* lab = labels + (size_t)b * Sc;
  const unsigned char* mk = masks + (size_t)b * Sc;
  const float* e = emit + (size_t)b * Sc * Lc;
  int prev = STARTc;
  float g = 0.0f;
  for (int s = 0; s < Sc; ++s) {
    if (!mk[s]) break;
    int l = lab[s];
    g += T[prev * Lc + l] + e[s * Lc + l];
    prev = l;
  }
  g += T[prev * Lc + PADc];
  gold[b] = g;
}

// ---------------------------------------------------------------------------
// Final scalar, deterministic fixed-order reduction (no float atomics).
// ---------------------------------------------------------------------------
__global__ void crf_reduce(const float* __restrict__ logZpart,
                           const float* __restrict__ gold,
                           float* __restrict__ out) {
  if (threadIdx.x == 0 && blockIdx.x == 0) {
    float z = 0.0f, g = 0.0f;
    for (int i = 0; i < Bc / 16; ++i) z += logZpart[i];
    for (int i = 0; i < Bc; ++i) g += gold[i];
    out[0] = (z - g) / (float)Bc;
  }
}

extern "C" void kernel_launch(void* const* d_in, const int* in_sizes, int n_in,
                              void* d_out, int out_size, void* d_ws, size_t ws_size,
                              hipStream_t stream) {
  const float*         emit   = (const float*)d_in[0];
  const int*           labels = (const int*)d_in[1];
  const unsigned char* masks  = (const unsigned char*)d_in[2];
  const float*         T      = (const float*)d_in[3];

  char* ws = (char*)d_ws;
  _Float16* expTh    = (_Float16*)(ws + 0);     // 64*48*2 = 6144 B
  float*    expTpad  = (float*)(ws + 6144);     // 48*4    =  192 B
  float*    logZpart = (float*)(ws + 6400);     // 32*4    =  128 B
  float*    gold     = (float*)(ws + 6656);     // 512*4   = 2048 B

  hipLaunchKernelGGL(crf_prep, dim3(1), dim3(256), 0, stream, T, expTh, expTpad);
  hipLaunchKernelGGL(crf_forward, dim3(Bc / 16), dim3(32), 0, stream,
                     emit, masks, T, expTh, expTpad, logZpart);
  hipLaunchKernelGGL(crf_gold, dim3(2), dim3(256), 0, stream,
                     emit, labels, masks, T, gold);
  hipLaunchKernelGGL(crf_reduce, dim3(1), dim3(64), 0, stream,
                     logZpart, gold, (float*)d_out);
}
